// MessageEncoderLongformer_38843684225571
// MI455X (gfx1250) — matrix operand order repaired and checked by reading, not compiled, not run on hardware
//
#include <hip/hip_runtime.h>
#include <math.h>

// ---------------- problem constants ----------------
#define B_   2
#define S_   4096
#define DM_  768
#define H_   12
#define HD_  64
#define G_   16
#define OUT_ 512
#define HP_  16            // heads padded to 16 for WMMA N-tile

// ---------------- CDNA5 WMMA types ----------------
typedef __attribute__((ext_vector_type(16))) __bf16 v16bf;
typedef __attribute__((ext_vector_type(8)))  float  v8f;
typedef int v4i __attribute__((ext_vector_type(4)));

union FragU { v16bf v; uint4 u[2]; unsigned us[8]; };

// CDNA5 async global->LDS path (guarded: falls back to the proven
// VGPR-staged transpose if the toolchain lacks the builtins).
#if __has_builtin(__builtin_amdgcn_global_load_async_to_lds_b128) && \
    __has_builtin(__builtin_amdgcn_s_wait_asynccnt)
#define USE_ASYNC_LDS 1
typedef __attribute__((address_space(1))) v4i as1_v4i;   // global-space 16B chunk
typedef __attribute__((address_space(3))) v4i as3_v4i;   // LDS-space 16B chunk
#endif

static __device__ inline unsigned short f2bf(float f) {
    unsigned u = __builtin_bit_cast(unsigned, f);
    unsigned r = u + 0x7FFFu + ((u >> 16) & 1u);   // round-to-nearest-even
    return (unsigned short)(r >> 16);
}

// A fragment: 16x32 bf16, row-major source with leading dim `ld` (elements).
// lane m = lane&15 selects row; half = lane>>4 selects K = 16h .. 16h+15 (contiguous).
static __device__ inline v16bf load_frag_a(const unsigned short* base, int ld, int k0, int lane) {
    int half = lane >> 4, m = lane & 15;
    const uint4* p = (const uint4*)(base + (size_t)m * ld + k0 + 16 * half);
    FragU f;
    f.u[0] = p[0];   // K = k0 + 16h .. +7
    f.u[1] = p[1];   // K = k0 + 16h + 8 .. +15
    return f.v;
}

// B fragment: 32x16 bf16 taken from a TRANSPOSED store Bt[n][k] (leading dim `ld`).
// lane n = lane&15 selects column; half = lane>>4 selects K = 16h .. 16h+15 (contiguous).
static __device__ inline v16bf load_frag_bt(const unsigned short* baseT, int ld, int k0, int lane) {
    int half = lane >> 4, n = lane & 15;
    const uint4* p = (const uint4*)(baseT + (size_t)n * ld + k0 + 16 * half);
    FragU f;
    f.u[0] = p[0];
    f.u[1] = p[1];
    return f.v;
}

static __device__ inline v8f wmma_bf16(v16bf a, v16bf b, v8f c) {
    // D = A*B + C, f32 accumulate, 16x16x32 bf16
    return __builtin_amdgcn_wmma_f32_16x16x32_bf16(false, a, false, b, (short)0, c, false, false);
}

// ---------------- K0: f32 -> bf16 convert of x ----------------
__global__ __launch_bounds__(256) void convert_kernel(const float* __restrict__ x,
                                                      unsigned short* __restrict__ xb, int n4) {
    int i = blockIdx.x * blockDim.x + threadIdx.x;
    if (i >= n4) return;
    float4 f = ((const float4*)x)[i];
    uint2 o;
    o.x = (unsigned)f2bf(f.x) | ((unsigned)f2bf(f.y) << 16);
    o.y = (unsigned)f2bf(f.z) | ((unsigned)f2bf(f.w) << 16);
    ((uint2*)xb)[i] = o;
}

// ---------------- generic small GEMV: out[b][j] = act((vec_b . W[:,j] + bias[j]) * scale) ----
__global__ __launch_bounds__(256) void gemv_kernel(const float* __restrict__ v, int vstride,
                                                   const float* __restrict__ W,
                                                   const float* __restrict__ bias,
                                                   float* __restrict__ out,
                                                   int K, int N, float scale, int doTanh) {
    int j = blockIdx.x * blockDim.x + threadIdx.x;
    int b = blockIdx.y;
    if (j >= N) return;
    const float* vec = v + (size_t)b * vstride;
    float acc = 0.f;
    for (int i = 0; i < K; ++i) acc = fmaf(vec[i], W[(size_t)i * N + j], acc);
    acc = (acc + bias[j]) * scale;
    out[(size_t)b * N + j] = doTanh ? tanhf(acc) : acc;
}

// ---------------- K1b: fold qg0 into wkg: uT[b][h][i] = sum_e wkg[i, h*64+e]*qg0[b,h*64+e]
// also c[b][h] = bkg_h . qg0_h. h padded to 16 (zeros).
__global__ __launch_bounds__(256) void build_u_kernel(const float* __restrict__ qg0,
                                                      const float* __restrict__ wkg,
                                                      const float* __restrict__ bkg,
                                                      unsigned short* __restrict__ uT,
                                                      float* __restrict__ cvec) {
    int b = blockIdx.y;
    int idx = blockIdx.x * 256 + threadIdx.x;     // 0 .. HP_*DM_-1
    int h = idx / DM_, i = idx % DM_;
    const float* q = qg0 + (size_t)b * DM_;
    float val = 0.f;
    if (h < H_) {
        const float* wrow = wkg + (size_t)i * DM_ + h * HD_;
        const float* qh = q + h * HD_;
        for (int e = 0; e < HD_; ++e) val = fmaf(wrow[e], qh[e], val);
    }
    uT[((size_t)b * HP_ + h) * DM_ + i] = f2bf(val);
    if (i == 0) {
        float c = 0.f;
        if (h < H_) {
            const float* qh = q + h * HD_;
            for (int e = 0; e < HD_; ++e) c = fmaf(bkg[h * HD_ + e], qh[e], c);
        }
        cvec[b * HP_ + h] = c;
    }
}

// ---------------- K2: scores[b][h][s] = x[b,s].u[b,h] + c[b,h]
// GEMM [S,768] (bf16) x [768,16] (bf16, pre-transposed) -> f32, WMMA 16x16x32.
__global__ __launch_bounds__(256) void score_kernel(const unsigned short* __restrict__ xb,
                                                    const unsigned short* __restrict__ uT,
                                                    const float* __restrict__ cvec,
                                                    float* __restrict__ scores) {
    int b = blockIdx.y;
    int lane = threadIdx.x & 31, wave = threadIdx.x >> 5;
    int sBase = (blockIdx.x * 8 + wave) * 16;                 // 256 M-tiles per batch
    const unsigned short* A  = xb + (size_t)b * S_ * DM_ + (size_t)sBase * DM_;
    const unsigned short* Bt = uT + (size_t)b * HP_ * DM_;
    v8f acc = {0.f, 0.f, 0.f, 0.f, 0.f, 0.f, 0.f, 0.f};
    for (int k0 = 0; k0 < DM_; k0 += 32) {
        if (k0 + 32 < DM_)
            __builtin_prefetch(A + (size_t)(lane & 15) * DM_ + k0 + 32, 0, 1);
        v16bf a  = load_frag_a(A, DM_, k0, lane);
        v16bf bb = load_frag_bt(Bt, DM_, k0, lane);
        acc = wmma_bf16(a, bb, acc);
    }
    int half = lane >> 4, n = lane & 15;
    float c = cvec[b * HP_ + n];
    float* orow = scores + ((size_t)b * HP_ + n) * S_;
    for (int j = 0; j < 8; ++j)
        orow[sBase + 8 * half + j] = acc[j] + c;
}

// ---------------- K3: softmax over s (4096) per (b,h); write normalized p as bf16 A-matrix
__global__ __launch_bounds__(256) void softmax_kernel(const float* __restrict__ scores,
                                                      unsigned short* __restrict__ p_bf) {
    int h = blockIdx.x, b = blockIdx.y, tid = threadIdx.x;
    unsigned short* prow = p_bf + ((size_t)b * HP_ + h) * S_;
    if (h >= H_) {                      // zero the padded A rows
        for (int i = tid; i < S_; i += 256) prow[i] = 0;
        return;
    }
    const float* s = scores + ((size_t)b * HP_ + h) * S_;
    __shared__ float red[256];
    float mx = -3.0e38f;
    for (int i = tid; i < S_; i += 256) mx = fmaxf(mx, s[i]);
    red[tid] = mx; __syncthreads();
    for (int off = 128; off > 0; off >>= 1) {
        if (tid < off) red[tid] = fmaxf(red[tid], red[tid + off]);
        __syncthreads();
    }
    float M = red[0]; __syncthreads();
    float sum = 0.f;
    for (int i = tid; i < S_; i += 256) sum += __expf(s[i] - M);
    red[tid] = sum; __syncthreads();
    for (int off = 128; off > 0; off >>= 1) {
        if (tid < off) red[tid] += red[tid + off];
        __syncthreads();
    }
    float inv = 1.f / red[0];
    for (int i = tid; i < S_; i += 256) prow[i] = f2bf(__expf(s[i] - M) * inv);
}

// ---------------- K4: Y[b][16][768] = P[16x4096] @ Xb[4096x768] (per batch)
// Async path: X tile copied global->LDS row-major via GLOBAL_LOAD_ASYNC_TO_LDS_B128
//             (no VGPR round trip, ASYNCcnt-tracked), B fragment gathered as bf16 pairs.
// Fallback:   X tile staged through LDS with a VGPR transpose, fragments as ds b128.
__global__ __launch_bounds__(256) void pv_kernel(const unsigned short* __restrict__ p_bf,
                                                 const unsigned short* __restrict__ xb,
                                                 float* __restrict__ Y) {
    int b = blockIdx.y;
    int tid = threadIdx.x, lane = tid & 31, wave = tid >> 5;
    int nBlockBase = blockIdx.x * 128;                 // 6 blocks cover N=768
#ifdef USE_ASYNC_LDS
    const int LDP = 132;                               // padded pitch (u16) to break bank conflicts
    __shared__ __attribute__((aligned(16))) unsigned short Ks[32 * LDP]; // row-major [k][n]
#else
    __shared__ __attribute__((aligned(16))) unsigned short Bs[128 * 32]; // [n][k] transposed, 8 KB
#endif
    const unsigned short* X = xb + (size_t)b * S_ * DM_;
    const unsigned short* A = p_bf + (size_t)b * HP_ * S_;
    v8f acc = {0.f, 0.f, 0.f, 0.f, 0.f, 0.f, 0.f, 0.f};
    for (int k0 = 0; k0 < S_; k0 += 32) {
        __syncthreads();
#ifdef USE_ASYNC_LDS
        // 32(k) x 128(n) bf16 tile: 512 x 16B chunks, copied straight into LDS.
        for (int it = 0; it < 2; ++it) {
            int idx = tid + it * 256;                  // 0..511
            int kr  = idx >> 4;                        // 0..31
            int c16 = idx & 15;                        // 0..15 (8 bf16 each)
            const unsigned short* gsrc = X + (size_t)(k0 + kr) * DM_ + nBlockBase + c16 * 8;
            if (k0 + 32 < S_) __builtin_prefetch(gsrc + 32 * DM_, 0, 1);
            unsigned short* ldst = &Ks[kr * LDP + c16 * 8];
            __builtin_amdgcn_global_load_async_to_lds_b128(
                (as1_v4i*)gsrc, (as3_v4i*)ldst, 0, 0);
        }
        __builtin_amdgcn_s_wait_asynccnt(0);
        __syncthreads();
        v16bf a = load_frag_a(A, S_, k0, lane);
        FragU f;
        {
            int n = lane & 15, half = lane >> 4;
            const unsigned short* col = &Ks[(16 * half) * LDP + (wave * 16 + n)];
#pragma unroll
            for (int j = 0; j < 8; ++j) {
                unsigned lo = col[(2 * j) * LDP];
                unsigned hi = col[(2 * j + 1) * LDP];
                f.us[j] = lo | (hi << 16);
            }
        }
        acc = wmma_bf16(a, f.v, acc);
#else
        // load 32(k) x 128(n) bf16 tile, store transposed into LDS
        for (int it = 0; it < 2; ++it) {
            int idx = tid + it * 256;                  // 0..511 uint4s
            int kr  = idx >> 4;                        // 0..31
            int c16 = idx & 15;                        // 0..15 (8 bf16 each)
            const unsigned short* gsrc = X + (size_t)(k0 + kr) * DM_ + nBlockBase + c16 * 8;
            if (k0 + 32 < S_) __builtin_prefetch(gsrc + 32 * DM_, 0, 1);
            uint4 val = *(const uint4*)gsrc;
            const unsigned short* vs = (const unsigned short*)&val;
            int nl = c16 * 8;
            for (int q = 0; q < 8; ++q) Bs[(nl + q) * 32 + kr] = vs[q];
        }
        __syncthreads();
        v16bf a  = load_frag_a(A, S_, k0, lane);
        v16bf bb = load_frag_bt(&Bs[(wave * 16) * 32], 32, 0, lane);
        acc = wmma_bf16(a, bb, acc);
#endif
    }
    int half = lane >> 4, n = lane & 15;
    int nGlob = nBlockBase + wave * 16 + n;
    for (int j = 0; j < 8; ++j)
        Y[((size_t)b * HP_ + 8 * half + j) * DM_ + nGlob] = acc[j];
}

// ---------------- K5: og[b][d] = Y[b][h(d)] . wvg[:,d] + bvg[d]
__global__ __launch_bounds__(256) void og_kernel(const float* __restrict__ Y,
                                                 const float* __restrict__ wvg,
                                                 const float* __restrict__ bvg,
                                                 float* __restrict__ og) {
    int b = blockIdx.y;
    int d = blockIdx.x * 256 + threadIdx.x;
    if (d >= DM_) return;
    int h = d >> 6;
    const float* y = Y + ((size_t)b * HP_ + h) * DM_;
    float acc = 0.f;
    for (int i = 0; i < DM_; ++i) acc = fmaf(y[i], wvg[(size_t)i * DM_ + d], acc);
    og[(size_t)b * DM_ + d] = acc + bvg[d];
}

// ---------------- host-side orchestration ----------------
extern "C" void kernel_launch(void* const* d_in, const int* in_sizes, int n_in,
                              void* d_out, int out_size, void* d_ws, size_t ws_size,
                              hipStream_t stream) {
    const float* x   = (const float*)d_in[0];
    const float* wqg = (const float*)d_in[7];
    const float* bqg = (const float*)d_in[8];
    const float* wkg = (const float*)d_in[9];
    const float* bkg = (const float*)d_in[10];
    const float* wvg = (const float*)d_in[11];
    const float* bvg = (const float*)d_in[12];
    const float* wo  = (const float*)d_in[13];
    const float* bo  = (const float*)d_in[14];
    const float* wp  = (const float*)d_in[15];
    const float* bp  = (const float*)d_in[16];
    const float* wfc = (const float*)d_in[17];
    const float* bfc = (const float*)d_in[18];
    float* out = (float*)d_out;

    char* ws = (char*)d_ws;
    size_t off = 0;
    auto alloc = [&](size_t bytes) -> char* {
        char* p = ws + off;
        off = (off + bytes + 255) & ~(size_t)255;
        return p;
    };
    unsigned short* xb     = (unsigned short*)alloc((size_t)B_ * S_ * DM_ * 2);   // 12.6 MB bf16 copy of x
    unsigned short* uT     = (unsigned short*)alloc((size_t)B_ * HP_ * DM_ * 2);  // folded wkg.qg0, transposed
    float*          qg0    = (float*)alloc((size_t)B_ * DM_ * 4);
    float*          cvec   = (float*)alloc((size_t)B_ * HP_ * 4);
    float*          scores = (float*)alloc((size_t)B_ * HP_ * S_ * 4);
    unsigned short* pbf    = (unsigned short*)alloc((size_t)B_ * HP_ * S_ * 2);
    float*          Y      = (float*)alloc((size_t)B_ * HP_ * DM_ * 4);
    float*          og     = (float*)alloc((size_t)B_ * DM_ * 4);
    float*          attn0  = (float*)alloc((size_t)B_ * DM_ * 4);
    float*          pooled = (float*)alloc((size_t)B_ * DM_ * 4);
    (void)ws_size; (void)in_sizes; (void)n_in; (void)out_size;

    // x -> bf16 (single pass; both WMMA stages read this copy, L2-resident at 12.6 MB)
    int n4 = (B_ * S_ * DM_) / 4;
    convert_kernel<<<dim3(n4 / 256), dim3(256), 0, stream>>>(x, xb, n4);
    // qg0 = (x[:,0] @ wqg + bqg) * 1/sqrt(HD)
    gemv_kernel<<<dim3(3, B_), dim3(256), 0, stream>>>(x, S_ * DM_, wqg, bqg, qg0,
                                                       DM_, DM_, 0.125f, 0);
    // fold qg0 into wkg -> uT (bf16, transposed), plus bias-dot c
    build_u_kernel<<<dim3((HP_ * DM_) / 256, B_), dim3(256), 0, stream>>>(qg0, wkg, bkg, uT, cvec);
    // scores[b,h,s] via WMMA GEMM [4096,768]x[768,16]
    score_kernel<<<dim3(32, B_), dim3(256), 0, stream>>>(xb, uT, cvec, scores);
    // softmax over s, emit normalized probabilities as bf16 A-matrix (heads padded to 16)
    softmax_kernel<<<dim3(HP_, B_), dim3(256), 0, stream>>>(scores, pbf);
    // Y = P^T X via WMMA GEMM [16,4096]x[4096,768]
    pv_kernel<<<dim3(6, B_), dim3(256), 0, stream>>>(pbf, xb, Y);
    // og = Y @ wvg (per-head slices) + bvg   (== attention output row 0)
    og_kernel<<<dim3(3, B_), dim3(256), 0, stream>>>(Y, wvg, bvg, og);
    // attn0 = og @ wo + bo ; pooled = tanh(attn0 @ wp + bp) ; out = pooled @ wfc + bfc
    gemv_kernel<<<dim3(3, B_), dim3(256), 0, stream>>>(og,    DM_, wo,  bo,  attn0, DM_, DM_,  1.f, 0);
    gemv_kernel<<<dim3(3, B_), dim3(256), 0, stream>>>(attn0, DM_, wp,  bp,  pooled, DM_, DM_, 1.f, 1);
    gemv_kernel<<<dim3(2, B_), dim3(256), 0, stream>>>(pooled, DM_, wfc, bfc, out,   DM_, OUT_, 1.f, 0);
}
